// my_LSTM_352187318676
// MI455X (gfx1250) — matrix operand order repaired and checked by reading, not compiled, not run on hardware
//
#include <hip/hip_runtime.h>
#include <math.h>

// Problem constants (L, N, H) from the reference: x:[64,1024,1024], H==N==1024.
#define NH 1024
#define NN (1024 * 1024)
#define LSTEPS 64

// GEMM tiling: block tile 128x64, 8 waves (wave32) in 4x2, wave tile 32x32.
#define BM 128
#define BN 64
#define BK 32
#define LDSTR 40  // bf16 elements per LDS row: 32 + 8 pad (dodges bank conflicts)

// Shared memory: GEMM staging (A + up to 2 B tiles) reused as epilogue
// staging buffer (BM x BN bf16) after the final barrier of the K-loop.
#define SMEM_ELTS (BM * LDSTR + 2 * BN * LDSTR)  // 10240 bf16 = 20 KB >= 16 KB stage

typedef __attribute__((ext_vector_type(16))) __bf16 v16bf;
typedef __attribute__((ext_vector_type(8)))  __bf16 v8bf;
typedef __attribute__((ext_vector_type(8)))  float  v8f;

// ---------------------------------------------------------------------------
// Branch-free transcendentals on CDNA5 TRANS units (v_exp/v_rcp/v_tanh).
// ---------------------------------------------------------------------------
__device__ __forceinline__ float fast_rcp(float x) {
#if __has_builtin(__builtin_amdgcn_rcpf)
    return __builtin_amdgcn_rcpf(x);
#else
    return 1.0f / x;
#endif
}

__device__ __forceinline__ float sigmoid_f(float x) {
    return fast_rcp(1.0f + __expf(-x));
}

__device__ __forceinline__ float tanh_f(float x) {
#if __has_builtin(__builtin_amdgcn_tanhf)
    return __builtin_amdgcn_tanhf(x);
#else
    // tanh(x) = 1 - 2/(exp(2x)+1); |x|>~9 saturates naturally in f32.
    float e = __expf(2.0f * x);
    return 1.0f - 2.0f * fast_rcp(e + 1.0f);
#endif
}

// ---------------------------------------------------------------------------
// Fragment loaders (wave32 CDNA5 WMMA 16x16x32 bf16 layouts, CDNA5 ISA)
// A 16x32 (MxK): lane&15 = M row; lanes 0-15 hold K {0..7,16..23},
//                lanes 16-31 hold K {8..15,24..31}; packed 2/VGPR in K order.
// ---------------------------------------------------------------------------
__device__ __forceinline__ v16bf load_frag_A(const __bf16* lds, int row, int lane) {
    const int kb = (lane >= 16) ? 8 : 0;
    const __bf16* p = lds + row * LDSTR + kb;
    v8bf lo = *(const v8bf*)(p);        // K = kb .. kb+7
    v8bf hi = *(const v8bf*)(p + 16);   // K = kb+16 .. kb+23
    return __builtin_shufflevector(lo, hi, 0, 1, 2, 3, 4, 5, 6, 7,
                                   8, 9, 10, 11, 12, 13, 14, 15);
}

// B 32x16 (KxN): lane&15 = N col; lanes 0-15 hold K {0..7,16..23}, lanes 16-31
// hold K {8..15,24..31} (same order as A). LDS stores B transposed: [n][k].
__device__ __forceinline__ v16bf load_frag_B(const __bf16* lds, int col, int lane) {
    const int kb = (lane >= 16) ? 8 : 0;
    const __bf16* p = lds + col * LDSTR + kb;
    v8bf lo = *(const v8bf*)(p);
    v8bf hi = *(const v8bf*)(p + 16);
    return __builtin_shufflevector(lo, hi, 0, 1, 2, 3, 4, 5, 6, 7,
                                   8, 9, 10, 11, 12, 13, 14, 15);
}

// ---------------------------------------------------------------------------
// Cooperative tile staging (256 threads)
// ---------------------------------------------------------------------------
template <bool A_IS_F32>
__device__ __forceinline__ void stage_A(__bf16* ldsA, const void* Ag,
                                        int m0, int k0, int tid) {
    if constexpr (A_IS_F32) {
        const float* A = (const float*)Ag;
#pragma unroll
        for (int it = 0; it < 4; ++it) {          // 128x32 f32 -> bf16
            int idx = it * 256 + tid;
            int row = idx >> 3, cg = (idx & 7) << 2;
            float4 v = *(const float4*)(A + (size_t)(m0 + row) * NH + k0 + cg);
            __bf16* d = ldsA + row * LDSTR + cg;
            d[0] = (__bf16)v.x; d[1] = (__bf16)v.y;
            d[2] = (__bf16)v.z; d[3] = (__bf16)v.w;
        }
    } else {
        const __bf16* A = (const __bf16*)Ag;
#pragma unroll
        for (int it = 0; it < 2; ++it) {          // 128x32 bf16, 16B chunks
            int idx = it * 256 + tid;
            int row = idx >> 2, cg = (idx & 3) << 3;
            *(v8bf*)(ldsA + row * LDSTR + cg) =
                *(const v8bf*)(A + (size_t)(m0 + row) * NH + k0 + cg);
        }
    }
}

// Stage B[k0:k0+32, n0:n0+64] transposed into LDS as [n][k] (coalesced reads).
template <bool B_IS_F32>
__device__ __forceinline__ void stage_B(__bf16* ldsB, const void* Bg,
                                        int k0, int n0, int tid) {
    if constexpr (B_IS_F32) {
        const float* B = (const float*)Bg;
#pragma unroll
        for (int it = 0; it < 2; ++it) {          // 32x64 f32
            int idx = it * 256 + tid;
            int kk = idx >> 4, ng = (idx & 15) << 2;
            float4 v = *(const float4*)(B + (size_t)(k0 + kk) * NH + n0 + ng);
            ldsB[(ng + 0) * LDSTR + kk] = (__bf16)v.x;
            ldsB[(ng + 1) * LDSTR + kk] = (__bf16)v.y;
            ldsB[(ng + 2) * LDSTR + kk] = (__bf16)v.z;
            ldsB[(ng + 3) * LDSTR + kk] = (__bf16)v.w;
        }
    } else {
        const __bf16* B = (const __bf16*)Bg;
        int idx = tid;                            // 32x64 bf16, one pass
        int kk = idx >> 3, ng = (idx & 7) << 3;
        v8bf v = *(const v8bf*)(B + (size_t)(k0 + kk) * NH + n0 + ng);
#pragma unroll
        for (int j = 0; j < 8; ++j) ldsB[(ng + j) * LDSTR + kk] = v[j];
    }
}

// ---------------------------------------------------------------------------
// GEMM core: accumulates up to two 1024-K GEMMs sharing the same A operand.
// Ends with a __syncthreads(), after which the LDS tiles may be reused.
// ---------------------------------------------------------------------------
template <int NOUT, bool A_IS_F32, bool B_IS_F32>
__device__ __forceinline__ void gemm_core(v8f (&acc)[NOUT][2][2],
                                          const void* Ag, const void* Bg0,
                                          const void* Bg1, __bf16* ldsA,
                                          __bf16* ldsB0, __bf16* ldsB1,
                                          int m0, int n0, int tid) {
    const int lane = tid & 31;
    const int wave = tid >> 5;
    const int wm = (wave >> 1) * 32;
    const int wn = (wave & 1) * 32;

    for (int k0 = 0; k0 < NH; k0 += BK) {
        stage_A<A_IS_F32>(ldsA, Ag, m0, k0, tid);
        stage_B<B_IS_F32>(ldsB0, Bg0, k0, n0, tid);
        if constexpr (NOUT == 2) stage_B<B_IS_F32>(ldsB1, Bg1, k0, n0, tid);
        __syncthreads();

        v16bf afr[2], bfr0[2], bfr1[2];
#pragma unroll
        for (int mi = 0; mi < 2; ++mi)
            afr[mi] = load_frag_A(ldsA, wm + mi * 16 + (lane & 15), lane);
#pragma unroll
        for (int ni = 0; ni < 2; ++ni) {
            bfr0[ni] = load_frag_B(ldsB0, wn + ni * 16 + (lane & 15), lane);
            if constexpr (NOUT == 2)
                bfr1[ni] = load_frag_B(ldsB1, wn + ni * 16 + (lane & 15), lane);
        }
#pragma unroll
        for (int mi = 0; mi < 2; ++mi) {
#pragma unroll
            for (int ni = 0; ni < 2; ++ni) {
                acc[0][mi][ni] = __builtin_amdgcn_wmma_f32_16x16x32_bf16(
                    false, afr[mi], false, bfr0[ni], (short)0, acc[0][mi][ni],
                    false, false);
                if constexpr (NOUT == 2)
                    acc[1][mi][ni] = __builtin_amdgcn_wmma_f32_16x16x32_bf16(
                        false, afr[mi], false, bfr1[ni], (short)0,
                        acc[1][mi][ni], false, false);
            }
        }
        __syncthreads();
    }
}

// C/D layout: lane&15 = N col; VGPR v = row M = v + (lane>=16 ? 8 : 0).
#define EPILOGUE_COORDS()                                                     \
    const int lane = tid & 31;                                                \
    const int wave = tid >> 5;                                                \
    const int wm = (wave >> 1) * 32, wn = (wave & 1) * 32;                    \
    const int rbase = wm + ((lane >= 16) ? 8 : 0);                            \
    const int cbase = wn + (lane & 15);

// Flush a staged BM x BN bf16 tile from LDS with coalesced b128 stores.
__device__ __forceinline__ void flush_stage_bf16(const __bf16* stg,
                                                 __bf16* __restrict__ gout,
                                                 int m0, int n0, int tid) {
#pragma unroll
    for (int it = 0; it < 4; ++it) {
        int i = it * 256 + tid;
        int row = i >> 3, cg = (i & 7) << 3;
        *(v8bf*)(gout + (size_t)(m0 + row) * NH + n0 + cg) =
            *(const v8bf*)(stg + row * BN + cg);
    }
}

// ---------------------------------------------------------------------------
// Phase 1 (one launch, blockIdx.z multiplex):
//   z==0: ah = x_t @ h                          -> bf16 (LDS-staged stores)
//   z==1: ig = sigmoid(h@Whi+bi)*tanh(h@Whg+bg) -> bf16 (LDS-staged stores)
// ---------------------------------------------------------------------------
__global__ __launch_bounds__(256) void k_phase1(
    const float* __restrict__ xt, const __bf16* __restrict__ hbf,
    const float* __restrict__ Whi, const float* __restrict__ Whg,
    const float* __restrict__ bi, const float* __restrict__ bg,
    __bf16* __restrict__ ahbf, __bf16* __restrict__ igbf) {
    __shared__ __bf16 smem[SMEM_ELTS];
    __bf16* ldsA  = smem;
    __bf16* ldsB0 = smem + BM * LDSTR;
    __bf16* ldsB1 = ldsB0 + BN * LDSTR;
    const int tid = threadIdx.x;
    const int m0 = blockIdx.y * BM, n0 = blockIdx.x * BN;

    if (blockIdx.z == 0) {
        v8f acc[1][2][2] = {};
        gemm_core<1, true, false>(acc, xt, hbf, nullptr, ldsA, ldsB0, nullptr,
                                  m0, n0, tid);
        EPILOGUE_COORDS();
#pragma unroll
        for (int mi = 0; mi < 2; ++mi)
#pragma unroll
            for (int ni = 0; ni < 2; ++ni)
#pragma unroll
                for (int v = 0; v < 8; ++v)
                    smem[(rbase + mi * 16 + v) * BN + cbase + ni * 16] =
                        (__bf16)acc[0][mi][ni][v];
        __syncthreads();
        flush_stage_bf16(smem, ahbf, m0, n0, tid);
    } else {
        v8f acc[2][2][2] = {};
        gemm_core<2, false, true>(acc, hbf, Whi, Whg, ldsA, ldsB0, ldsB1,
                                  m0, n0, tid);
        EPILOGUE_COORDS();
#pragma unroll
        for (int mi = 0; mi < 2; ++mi)
#pragma unroll
            for (int ni = 0; ni < 2; ++ni) {
                const int gcol = n0 + cbase + ni * 16;
                const float biv = bi[gcol], bgv = bg[gcol];
#pragma unroll
                for (int v = 0; v < 8; ++v) {
                    float iv = sigmoid_f(acc[0][mi][ni][v] + biv);
                    float gv = tanh_f(acc[1][mi][ni][v] + bgv);
                    smem[(rbase + mi * 16 + v) * BN + cbase + ni * 16] =
                        (__bf16)(iv * gv);
                }
            }
        __syncthreads();
        flush_stage_bf16(smem, igbf, m0, n0, tid);
    }
}

// ---------------------------------------------------------------------------
// Phase 2: f = sigmoid(ah@Whf+bf), o = sigmoid(ah@Who+bo) -> f32
// ---------------------------------------------------------------------------
__global__ __launch_bounds__(256) void k_fo(const __bf16* __restrict__ ahbf,
                                            const float* __restrict__ Whf,
                                            const float* __restrict__ Who,
                                            const float* __restrict__ bfv,
                                            const float* __restrict__ bov,
                                            float* __restrict__ fbuf,
                                            float* __restrict__ obuf) {
    __shared__ __bf16 smem[SMEM_ELTS];
    __bf16* ldsA  = smem;
    __bf16* ldsB0 = smem + BM * LDSTR;
    __bf16* ldsB1 = ldsB0 + BN * LDSTR;
    const int tid = threadIdx.x;
    const int m0 = blockIdx.y * BM, n0 = blockIdx.x * BN;
    v8f acc[2][2][2] = {};
    gemm_core<2, false, true>(acc, ahbf, Whf, Who, ldsA, ldsB0, ldsB1,
                              m0, n0, tid);
    EPILOGUE_COORDS();
#pragma unroll
    for (int mi = 0; mi < 2; ++mi)
#pragma unroll
        for (int ni = 0; ni < 2; ++ni) {
            const int gcol = n0 + cbase + ni * 16;
            const float bfb = bfv[gcol], bob = bov[gcol];
#pragma unroll
            for (int v = 0; v < 8; ++v) {
                const size_t idx = (size_t)(m0 + rbase + mi * 16 + v) * NH + gcol;
                fbuf[idx] = sigmoid_f(acc[0][mi][ni][v] + bfb);
                obuf[idx] = sigmoid_f(acc[1][mi][ni][v] + bob);
            }
        }
}

// ---------------------------------------------------------------------------
// Phase 3: cc = x_t @ ig; c = f*c + cc; h = o*tanh(c); emit h (f32 + bf16)
// ---------------------------------------------------------------------------
__global__ __launch_bounds__(256) void k_up(const float* __restrict__ xt,
                                            const __bf16* __restrict__ igbf,
                                            const float* __restrict__ fbuf,
                                            const float* __restrict__ obuf,
                                            float* __restrict__ cbuf,
                                            float* __restrict__ hout,
                                            __bf16* __restrict__ hbf) {
    __shared__ __bf16 smem[SMEM_ELTS];
    __bf16* ldsA  = smem;
    __bf16* ldsB0 = smem + BM * LDSTR;
    const int tid = threadIdx.x;
    const int m0 = blockIdx.y * BM, n0 = blockIdx.x * BN;
    v8f acc[1][2][2] = {};
    gemm_core<1, true, false>(acc, xt, igbf, nullptr, ldsA, ldsB0, nullptr,
                              m0, n0, tid);
    EPILOGUE_COORDS();
#pragma unroll
    for (int mi = 0; mi < 2; ++mi)
#pragma unroll
        for (int ni = 0; ni < 2; ++ni) {
            const int gcol = n0 + cbase + ni * 16;
#pragma unroll
            for (int v = 0; v < 8; ++v) {
                const int grow = m0 + rbase + mi * 16 + v;
                const size_t idx = (size_t)grow * NH + gcol;
                float cn = fbuf[idx] * cbuf[idx] + acc[0][mi][ni][v];
                cbuf[idx] = cn;
                float hv = obuf[idx] * tanh_f(cn);
                hout[idx] = hv;
                // also stage bf16 h for next step's GEMMs
                smem[(rbase + mi * 16 + v) * BN + cbase + ni * 16] = (__bf16)hv;
            }
        }
    __syncthreads();
    flush_stage_bf16(smem, hbf, m0, n0, tid);
}

__global__ void k_init(float* __restrict__ cbuf, __bf16* __restrict__ hbf,
                       float* __restrict__ out0) {
    int i = blockIdx.x * blockDim.x + threadIdx.x;
    if (i < NN) {
        cbuf[i] = 0.0f;
        hbf[i] = (__bf16)0.0f;
        out0[i] = 0.0f;   // out[0] = h0 = zeros
    }
}

// Tuple tail: h_t = out[L], c_t = c.
__global__ void k_tail(float* __restrict__ out, const float* __restrict__ cbuf) {
    int i = blockIdx.x * blockDim.x + threadIdx.x;
    if (i < NN) {
        out[(size_t)(LSTEPS + 1) * NN + i] = out[(size_t)LSTEPS * NN + i];
        out[(size_t)(LSTEPS + 2) * NN + i] = cbuf[i];
    }
}

// ---------------------------------------------------------------------------
extern "C" void kernel_launch(void* const* d_in, const int* in_sizes, int n_in,
                              void* d_out, int out_size, void* d_ws,
                              size_t ws_size, hipStream_t stream) {
    const float* x   = (const float*)d_in[0];
    const float* Whi = (const float*)d_in[1];
    const float* bi  = (const float*)d_in[2];
    const float* Whf = (const float*)d_in[3];
    const float* bfv = (const float*)d_in[4];
    const float* Whg = (const float*)d_in[5];
    const float* bg  = (const float*)d_in[6];
    const float* Who = (const float*)d_in[7];
    const float* bo  = (const float*)d_in[8];
    float* out = (float*)d_out;

    // Workspace carve: 3 f32 buffers + 3 bf16 buffers = 18 MB.
    float*  cbuf = (float*)d_ws;
    float*  fbuf = cbuf + NN;
    float*  obuf = fbuf + NN;
    __bf16* hbf  = (__bf16*)(obuf + NN);
    __bf16* ahbf = hbf + NN;
    __bf16* igbf = ahbf + NN;

    const dim3 grid2(NH / BN, NH / BM);     // 16 x 8 = 128 blocks
    const dim3 grid1(NH / BN, NH / BM, 2);  // phase-1 multiplex: ah + ig
    const dim3 blk(256);
    const int eltBlocks = (NN + 255) / 256;

    k_init<<<eltBlocks, 256, 0, stream>>>(cbuf, hbf, out);

    for (int t = 0; t < LSTEPS; ++t) {
        const float* xt = x + (size_t)t * NN;
        k_phase1<<<grid1, blk, 0, stream>>>(xt, hbf, Whi, Whg, bi, bg,
                                            ahbf, igbf);
        k_fo<<<grid2, blk, 0, stream>>>(ahbf, Whf, Who, bfv, bo, fbuf, obuf);
        k_up<<<grid2, blk, 0, stream>>>(xt, igbf, fbuf, obuf, cbuf,
                                        out + (size_t)(t + 1) * NN, hbf);
    }

    k_tail<<<eltBlocks, 256, 0, stream>>>(out, cbuf);
}
